// StackedPyrHourGlassLoss_15857019257543
// MI455X (gfx1250) — matrix operand order repaired and checked by reading, not compiled, not run on hardware
//
#include <hip/hip_runtime.h>
#include <hip/hip_bf16.h>

typedef __attribute__((ext_vector_type(16))) _Float16 v16h;
typedef __attribute__((ext_vector_type(8)))  float    v8f;
typedef __attribute__((ext_vector_type(2)))  int      v2i;

#define EMBED   32
#define NPATCH  1125          // 5*15*15
#define NPAD    1136          // 71 * 16
#define COLT    (NPAD / 16)   // 71 column tiles
#define EPS_D   1e-6f

// ---------------------------------------------------------------------------
// compile-time level configs (all divisions/modulos constant-fold)
// ---------------------------------------------------------------------------
template <int LVL> struct LCfg;
template <> struct LCfg<0> {
  static constexpr int D=8, H=160, W=160, nb0=4, nb1=17, nb2=17,
    s1=6, s2=6, o0=2, o1=30, o2=30, f1=4, f2=4, bndBase=4,
    Btotal=2312, Bpad=2320;
};
template <> struct LCfg<1> {
  static constexpr int D=8, H=80, W=80, nb0=4, nb1=9, nb2=9,
    s1=6, s2=6, o0=2, o1=15, o2=15, f1=2, f2=2, bndBase=0,
    Btotal=648, Bpad=656;
};
template <> struct LCfg<2> {
  static constexpr int D=8, H=40, W=40, nb0=4, nb1=9, nb2=9,
    s1=3, s2=3, o0=2, o1=7, o2=7, f1=1, f2=1, bndBase=0,
    Btotal=648, Bpad=656;
};

// ---------------------------------------------------------------------------
// 0) zero the 6 dice accumulators (num,den per level)
// ---------------------------------------------------------------------------
__global__ void zero_acc_kernel(float* acc) {
  if (threadIdx.x < 6) acc[threadIdx.x] = 0.f;
}

// ---------------------------------------------------------------------------
// 1) pack W (32 x 1125, row-major f32) -> Wpack (NPAD cols x 32 K, f16,
//    column-major: Wpack[n*32 + k]) so a WMMA B-fragment lane load is two
//    runs of 8 contiguous halves. Padded columns are zero.
// ---------------------------------------------------------------------------
__global__ __launch_bounds__(256) void pack_w_kernel(const float* __restrict__ Wm,
                                                     _Float16* __restrict__ wp) {
  int idx = blockIdx.x * blockDim.x + threadIdx.x;
  if (idx >= NPAD * EMBED) return;
  int n = idx >> 5;   // output column (0..NPAD-1)
  int k = idx & 31;   // K index
  float v = (n < NPATCH) ? Wm[k * NPATCH + n] : 0.f;
  wp[n * EMBED + k] = (_Float16)v;
}

// ---------------------------------------------------------------------------
// 2) per-patch: gather 32-ch embedding at the patch center (f16 row of the
//    WMMA A matrix), plus center gt id and validity flag.
//    One wave32 per (padded) patch; lane == channel.
// ---------------------------------------------------------------------------
template <int LVL>
__global__ __launch_bounds__(256) void build_emb_kernel(
    const float* __restrict__ pred, const int* __restrict__ tgt,
    _Float16* __restrict__ embA, int* __restrict__ centers,
    float* __restrict__ validf)
{
  using C = LCfg<LVL>;
  int gtid = blockIdx.x * blockDim.x + threadIdx.x;
  int row  = gtid >> 5;
  int lane = gtid & 31;
  if (row >= C::Bpad) return;
  if (row >= C::Btotal) {                    // padding rows: zero everything
    embA[row * EMBED + lane] = (_Float16)0.f;
    if (lane == 0) { centers[row] = 0; validf[row] = 0.f; }
    return;
  }
  constexpr int P  = C::nb0 * C::nb1 * C::nb2;
  constexpr long HW = (long)C::H * C::W;
  int n    = row / P;
  int rem  = row - n * P;
  int pd   = rem / (C::nb1 * C::nb2);
  int rem2 = rem - pd * (C::nb1 * C::nb2);
  int ph   = rem2 / C::nb2;
  int pw   = rem2 - ph * C::nb2;
  int cd = C::o0 + pd;            // stride0 == 1 at every level
  int ch = C::o1 + ph * C::s1;
  int cw = C::o2 + pw * C::s2;
  long spat = (long)cd * HW + (long)ch * C::W + cw;

  // pred layout: (N, 32, D, H, W)
  embA[row * EMBED + lane] =
      (_Float16)pred[(long)(n * EMBED + lane) * C::D * HW + spat];

  if (lane == 0) {
    // tgt layout: (N, 9, D, H, W); channel 0 = gt ids, 1..8 = affinities
    int center = tgt[(long)(n * 9) * C::D * HW + spat];
    int mina = 1;
    #pragma unroll
    for (int j = 0; j < 4; ++j) {
      int a = tgt[(long)(n * 9 + 1 + C::bndBase + j) * C::D * HW + spat];
      mina = (a < mina) ? a : mina;
    }
    centers[row] = center;
    // valid = (center != 0) && (on_bnd != 1) ; on_bnd==1 iff min(aff)==0
    validf[row]  = (center != 0 && mina != 0) ? 1.f : 0.f;
  }
}

// ---------------------------------------------------------------------------
// 3) fused WMMA GEMM (16x16 tile per wave, single K=32 v_wmma) +
//    sigmoid + window max-pool of (gt != center)/(gt == 0) + dice partials
// ---------------------------------------------------------------------------
template <int LVL>
__global__ __launch_bounds__(256) void wmma_dice_kernel(
    const _Float16* __restrict__ embA, const _Float16* __restrict__ wpack,
    const float* __restrict__ bias,
    const int* __restrict__ centers, const float* __restrict__ validf,
    const int* __restrict__ tgt,
    float* __restrict__ acc /* [num, den] for this level */)
{
  using C = LCfg<LVL>;
  constexpr int rowTiles = C::Bpad / 16;
  int wave = (blockIdx.x * blockDim.x + threadIdx.x) >> 5;
  if (wave >= rowTiles * COLT) return;     // wave-uniform: EXEC stays all-1s
  int lane    = threadIdx.x & 31;
  int tr      = wave / COLT;
  int tc      = wave - tr * COLT;
  int halfSel = lane >> 4;                 // 0: lanes 0-15, 1: lanes 16-31
  int l15     = lane & 15;

  // ---- A fragment (16-bit A 16x32 layout, ISA 7.12.2):
  //  lanes 0-15 : row=l15, K = {0..7, 16..23}
  //  lanes 16-31: row=l15, K = {8..15, 24..31}
  int arow = tr * 16 + l15;
  int akb  = halfSel * 8;
  const _Float16* ap = embA + arow * EMBED;
  v16h a = {};
  #pragma unroll
  for (int h = 0; h < 8; ++h) a[h]     = ap[akb + h];
  #pragma unroll
  for (int h = 0; h < 8; ++h) a[8 + h] = ap[akb + 16 + h];

  // ---- B fragment: lanes 0-15 hold K={0..7,16..23} of column l15, lanes 16-31 K={8..15,24..31}
  int bcol = tc * 16 + l15;
  int bkb  = halfSel * 8;
  const _Float16* bp = wpack + bcol * EMBED + bkb;
  v16h b = {};
  #pragma unroll
  for (int h = 0; h < 16; ++h) b[h] = bp[(h < 8) ? h : (h + 8)];

  v8f c = {};
  c = __builtin_amdgcn_wmma_f32_16x16x32_f16(
      /*neg_a=*/false, a, /*neg_b=*/false, b,
      /*c_mod=*/(short)0, c, /*reuse_a=*/false, /*reuse_b=*/false);

  // C/D layout: c[r] -> (M = tr*16 + r + 8*halfSel, N = tc*16 + l15)
  constexpr long HW = (long)C::H * C::W;
  constexpr int  P  = C::nb0 * C::nb1 * C::nb2;
  int  ncol  = tc * 16 + l15;
  bool colOK = (ncol < NPATCH);
  float bb   = colOK ? bias[ncol] : 0.f;
  int od = ncol / 225;           // 15*15 (constant divide)
  int rr = ncol - od * 225;
  int oh = rr / 15;
  int ow = rr - oh * 15;

  float num = 0.f, den = 0.f;
  #pragma unroll
  for (int r = 0; r < 8; ++r) {
    int m = tr * 16 + r + halfSel * 8;
    if (colOK && m < C::Btotal) {
      float v = validf[m];
      if (v != 0.f) {            // v == 1 here; invalid patches contribute 0
        float pp = 1.f / (1.f + __expf(-(c[r] + bb)));
        int center = centers[m];
        int n    = m / P;
        int rem  = m - n * P;
        int pd   = rem / (C::nb1 * C::nb2);
        int rem2 = rem - pd * (C::nb1 * C::nb2);
        int ph   = rem2 / C::nb2;
        int pw   = rem2 - ph * C::nb2;
        int gd  = pd + od;                 // dws0 == stride0 == 1
        int gh0 = ph * C::s1 + oh * C::f1;
        int gw0 = pw * C::s2 + ow * C::f2; // even whenever f2>1 (8B aligned)
        const int* gptr = tgt + ((long)(n * 9) * C::D + gd) * HW;
        bool me = false, ign = false;
        #pragma unroll
        for (int yy = 0; yy < C::f1; ++yy) {
          const int* rowp = gptr + (long)(gh0 + yy) * C::W + gw0;
          if constexpr (C::f2 == 1) {
            int g = rowp[0];
            me  |= (g != center);
            ign |= (g == 0);
          } else {
            #pragma unroll
            for (int x2 = 0; x2 < C::f2 / 2; ++x2) {
              v2i g2 = *(const v2i*)(rowp + 2 * x2);   // global_load_b64
              me  |= (g2.x != center) || (g2.y != center);
              ign |= (g2.x == 0)      || (g2.y == 0);
            }
          }
        }
        float trg = (!ign && me) ? 1.f : 0.f;
        float x   = ign ? 0.f : pp;
        num += x * trg;
        den += x * x + trg * trg;
      }
    }
  }

  // wave32 butterfly reduction, then one atomic pair per wave
  #pragma unroll
  for (int o = 16; o > 0; o >>= 1) {
    num += __shfl_xor(num, o, 32);
    den += __shfl_xor(den, o, 32);
  }
  if (lane == 0) {
    atomicAdd(&acc[0], num);
    atomicAdd(&acc[1], den);
  }
}

// ---------------------------------------------------------------------------
// 4) finalize: loss = sum_l -2*num_l / max(den_l, EPS)
// ---------------------------------------------------------------------------
__global__ void finalize_kernel(const float* __restrict__ acc,
                                float* __restrict__ out) {
  if (threadIdx.x == 0) {
    float loss = 0.f;
    #pragma unroll
    for (int l = 0; l < 3; ++l) {
      float n = acc[2 * l], d = acc[2 * l + 1];
      loss += -2.f * n / fmaxf(d, EPS_D);
    }
    out[0] = loss;
  }
}

// ---------------------------------------------------------------------------
template <int LVL>
static void launch_level(const float* pred, const int* tgt, const float* Wm,
                         const float* bias, _Float16* embA, int* centers,
                         float* validf, _Float16* wpack, float* acc,
                         hipStream_t stream) {
  using C = LCfg<LVL>;
  pack_w_kernel<<<(NPAD * EMBED + 255) / 256, 256, 0, stream>>>(Wm, wpack);
  build_emb_kernel<LVL><<<(C::Bpad * 32 + 255) / 256, 256, 0, stream>>>(
      pred, tgt, embA, centers, validf);
  constexpr int tiles  = (C::Bpad / 16) * COLT;
  constexpr int blocks = (tiles + 7) / 8;   // 8 waves (256 threads) per block
  wmma_dice_kernel<LVL><<<blocks, 256, 0, stream>>>(
      embA, wpack, bias, centers, validf, tgt, acc);
}

extern "C" void kernel_launch(void* const* d_in, const int* in_sizes, int n_in,
                              void* d_out, int out_size, void* d_ws, size_t ws_size,
                              hipStream_t stream) {
  (void)in_sizes; (void)n_in; (void)out_size; (void)ws_size;
  // setup_inputs dict order: pred0,tgt0,W0,b0, pred1,tgt1,W1,b1, pred2,tgt2,W2,b2
  const float* pred[3] = {(const float*)d_in[0], (const float*)d_in[4], (const float*)d_in[8]};
  const int*   tgt [3] = {(const int*)  d_in[1], (const int*)  d_in[5], (const int*)  d_in[9]};
  const float* Wm  [3] = {(const float*)d_in[2], (const float*)d_in[6], (const float*)d_in[10]};
  const float* bs  [3] = {(const float*)d_in[3], (const float*)d_in[7], (const float*)d_in[11]};
  const int Bpad[3] = {2320, 656, 656};

  // carve workspace (host-side pointer math only; graph-capture safe)
  char* ws = (char*)d_ws;
  size_t off = 0;
  auto alloc = [&](size_t bytes) -> void* {
    void* p = ws + off;
    off = (off + bytes + 255) & ~(size_t)255;
    return p;
  };
  float*    acc = (float*)alloc(6 * sizeof(float));
  _Float16* embA[3]; int* centers[3]; float* validf[3]; _Float16* wpack[3];
  for (int l = 0; l < 3; ++l) {
    embA[l]    = (_Float16*)alloc((size_t)Bpad[l] * EMBED * sizeof(_Float16));
    centers[l] = (int*)     alloc((size_t)Bpad[l] * sizeof(int));
    validf[l]  = (float*)   alloc((size_t)Bpad[l] * sizeof(float));
    wpack[l]   = (_Float16*)alloc((size_t)NPAD * EMBED * sizeof(_Float16));
  }

  zero_acc_kernel<<<1, 32, 0, stream>>>(acc);

  launch_level<0>(pred[0], tgt[0], Wm[0], bs[0], embA[0], centers[0],
                  validf[0], wpack[0], acc + 0, stream);
  launch_level<1>(pred[1], tgt[1], Wm[1], bs[1], embA[1], centers[1],
                  validf[1], wpack[1], acc + 2, stream);
  launch_level<2>(pred[2], tgt[2], Wm[2], bs[2], embA[2], centers[2],
                  validf[2], wpack[2], acc + 4, stream);

  finalize_kernel<<<1, 32, 0, stream>>>(acc, (float*)d_out);
}
